// KGEdges_45603962749244
// MI455X (gfx1250) — compile-verified, matrix operand order checked by reading, not run on hardware
//
#include <hip/hip_runtime.h>
#include <hip/hip_bf16.h>

#define BS 8
#define SL 256
#define ENC_DIM 1024
#define EDGE_DIM 256
#define MINUS_INF (-1.0e8f)

typedef __attribute__((ext_vector_type(16))) __bf16 v16bf;
typedef __attribute__((ext_vector_type(8)))  float  v8f;

// hardware tanh if the gfx1250 builtin is declared, else ocml fallback
#if defined(__has_builtin)
#if __has_builtin(__builtin_amdgcn_tanhf)
#define KG_TANH(x) __builtin_amdgcn_tanhf(x)
#endif
#endif
#ifndef KG_TANH
#define KG_TANH(x) tanhf(x)
#endif

// ---------------------------------------------------------------------------
// Kernel 1 (fused head+child GEMM):
//   Y(2048x256) = X(2048x1024) * W(256x1024)^T (+bias)
// blockIdx.y selects (Wh, bh)->head vs (Wc, 0)->child.
// One wave computes a 16x32 tile of Y: one shared A fragment, two B fragments,
// two WMMAs per K-step (A reuse halves A loads + bf16 conversions).
// Tiles: M = 2048/16 = 128, Npairs = 256/32 = 8 -> 1024 waves -> 128 blocks.
// ---------------------------------------------------------------------------
__global__ __launch_bounds__(256) void kg_gemm_wmma(
    const float* __restrict__ X,
    const float* __restrict__ Wh, const float* __restrict__ bh,
    const float* __restrict__ Wc,
    float* __restrict__ head, float* __restrict__ child) {
  const float* W    = (blockIdx.y == 0) ? Wh : Wc;
  const float* bias = (blockIdx.y == 0) ? bh : nullptr;
  float*       Y    = (blockIdx.y == 0) ? head : child;

  const int lane    = threadIdx.x & 31;
  const int wave    = threadIdx.x >> 5;
  const int id      = blockIdx.x * 8 + wave;
  const int tile_n2 = id & 7;    // N pair index: cols [tile_n2*32, +32)
  const int tile_m  = id >> 3;   // 128 M tiles
  const int half    = lane >> 4; // upper/lower half-wave (K split per ISA layout)
  const int l       = lane & 15;

  const float* xrow  = X + (size_t)(tile_m * 16 + l) * ENC_DIM;       // A row m = l
  const float* wrow0 = W + (size_t)(tile_n2 * 32 + l) * ENC_DIM;      // B0 col n = l
  const float* wrow1 = wrow0 + (size_t)16 * ENC_DIM;                  // B1 col n = l

  v8f c0 = {};
  v8f c1 = {};
  for (int kb = 0; kb < ENC_DIM; kb += 32) {
    v16bf a, b0, b1;
#pragma unroll
    for (int p = 0; p < 8; ++p) {
      // 16-bit A 16x32 layout: VGPR p holds K pair k0,k0+1;
      // k0 = 2*(p%4) + 16*(p/4) + 8*half  (half-wave K split)
      const int k0 = kb + 2 * (p & 3) + ((p >> 2) << 4) + (half << 3);
      const float2 av  = *(const float2*)(xrow + k0);
      const float2 bv0 = *(const float2*)(wrow0 + k0);
      const float2 bv1 = *(const float2*)(wrow1 + k0);
      a[2 * p]      = (__bf16)av.x;
      a[2 * p + 1]  = (__bf16)av.y;
      b0[2 * p]     = (__bf16)bv0.x;
      b0[2 * p + 1] = (__bf16)bv0.y;
      b1[2 * p]     = (__bf16)bv1.x;
      b1[2 * p + 1] = (__bf16)bv1.y;
    }
    // (neg_a, A, neg_b, B, c_mod, C, reuse_a, reuse_b)
    c0 = __builtin_amdgcn_wmma_f32_16x16x32_bf16(false, a, false, b0, (short)0,
                                                 c0, false, false);
    c1 = __builtin_amdgcn_wmma_f32_16x16x32_bf16(false, a, false, b1, (short)0,
                                                 c1, false, false);
  }

  const int   col0 = tile_n2 * 32 + l;
  const int   col1 = col0 + 16;
  const float ba0  = bias ? bias[col0] : 0.0f;
  const float ba1  = bias ? bias[col1] : 0.0f;
#pragma unroll
  for (int v = 0; v < 8; ++v) {
    // C/D layout: lanes 0-15 -> M=v, lanes 16-31 -> M=v+8, N = l
    const int row = tile_m * 16 + (half ? v + 8 : v);
    Y[(size_t)row * EDGE_DIM + col0] = c0[v] + ba0;
    Y[(size_t)row * EDGE_DIM + col1] = c1[v] + ba1;
  }
}

// ---------------------------------------------------------------------------
// Kernel 2: edge[b,i,j] = sum_d tanh(head[b,j,d] + child[b,i,d]) * w_out[d]
//           + minus_mask[b,i] + minus_mask[b,j]
// Block = 256 threads handles a 16x16 (i,j) tile for one batch b.
// child/head 16x256 slices staged in LDS, row stride 260 floats:
//   - rows stay 16B aligned (1040 B) -> ds_load_b128
//   - bank = (4*jl + d) mod 64 -> conflict-free across the 16 j rows
// ---------------------------------------------------------------------------
#define PSTRIDE 260

__global__ __launch_bounds__(256) void kg_pair(
    const float* __restrict__ head, const float* __restrict__ child,
    const float* __restrict__ w_out, const long long* __restrict__ mask,
    float* __restrict__ out) {
  __shared__ float sch[16][PSTRIDE];
  __shared__ float shd[16][PSTRIDE];
  __shared__ float sw[EDGE_DIM];

  const int t  = threadIdx.x;        // 0..255 -> d index for loads
  const int b  = blockIdx.z;
  const int i0 = blockIdx.y * 16;
  const int j0 = blockIdx.x * 16;

  sw[t] = w_out[t];
#pragma unroll
  for (int r = 0; r < 16; ++r) {     // coalesced: 256 consecutive floats per row
    sch[r][t] = child[((size_t)b * SL + (i0 + r)) * EDGE_DIM + t];
    shd[r][t] = head [((size_t)b * SL + (j0 + r)) * EDGE_DIM + t];
  }
  __syncthreads();

  const int il = t >> 4;
  const int jl = t & 15;

  float acc = 0.0f;
#pragma unroll 4
  for (int d = 0; d < EDGE_DIM; d += 4) {
    const float4 cv = *(const float4*)&sch[il][d];
    const float4 hv = *(const float4*)&shd[jl][d];
    const float4 wv = *(const float4*)&sw[d];
    acc = fmaf(KG_TANH(cv.x + hv.x), wv.x, acc);
    acc = fmaf(KG_TANH(cv.y + hv.y), wv.y, acc);
    acc = fmaf(KG_TANH(cv.z + hv.z), wv.z, acc);
    acc = fmaf(KG_TANH(cv.w + hv.w), wv.w, acc);
  }

  const float mi = (1.0f - (float)mask[b * SL + i0 + il]) * MINUS_INF;
  const float mj = (1.0f - (float)mask[b * SL + j0 + jl]) * MINUS_INF;
  out[((size_t)b * SL + (i0 + il)) * SL + (j0 + jl)] = acc + mi + mj;
}

// ---------------------------------------------------------------------------
extern "C" void kernel_launch(void* const* d_in, const int* in_sizes, int n_in,
                              void* d_out, int out_size, void* d_ws, size_t ws_size,
                              hipStream_t stream) {
  const float*     enc   = (const float*)d_in[0];      // (8,256,1024) f32
  const long long* mask  = (const long long*)d_in[1];  // (8,256) i64
  const float*     Wh    = (const float*)d_in[2];      // (256,1024) f32
  const float*     bh    = (const float*)d_in[3];      // (256,) f32
  const float*     Wc    = (const float*)d_in[4];      // (256,1024) f32
  const float*     w_out = (const float*)d_in[5];      // (256,) f32
  float*           out   = (float*)d_out;              // (8,256,256) f32

  float* head  = (float*)d_ws;                         // 2 MB
  float* child = head + (size_t)BS * SL * EDGE_DIM;    // 2 MB

  // 1024 16x32 wave-tiles per projection, 8 waves/block -> grid (128, 2)
  kg_gemm_wmma<<<dim3(128, 2), dim3(256), 0, stream>>>(enc, Wh, bh, Wc, head,
                                                       child);

  kg_pair<<<dim3(SL / 16, SL / 16, BS), dim3(256), 0, stream>>>(head, child,
                                                                w_out, mask, out);
}
